// AtrousAttentionBlock_62775241998674
// MI455X (gfx1250) — matrix operand order repaired and checked by reading, not compiled, not run on hardware
//
#include <hip/hip_runtime.h>
#include <hip/hip_bf16.h>

typedef __attribute__((ext_vector_type(16))) _Float16 v16h;
typedef __attribute__((ext_vector_type(8)))  _Float16 v8h;
typedef __attribute__((ext_vector_type(8)))  float    v8f;

#define IMGS 4
#define HH 128
#define WW 128
#define HW 16384
#define NPIX 65536
#define CCH 256
#define DFC 8
#define NDB 16
#define CATCH 384
#define EPSLN 1e-3f

#define LDA 72   // halves; 144B row stride (16B aligned)
#define LDB 72

// ---------------- WMMA fragment helpers (CDNA5 16x16x32 f16, wave32) ----------
// A (16x32 f16): lane L<16 -> row M=L, K = kb..kb+7 and kb+16..kb+23 (kb=0);
// lanes 16..31 -> kb=8. Two contiguous 8-half runs => 2x ds_load_b128.
__device__ __forceinline__ v16h frag_a(const _Float16* sA) {
  int lane = threadIdx.x & 31;
  const _Float16* row = sA + (lane & 15) * LDA + ((lane >> 4) << 3);
  v8h lo = *(const v8h*)(row);
  v8h hi = *(const v8h*)(row + 16);
  return __builtin_shufflevector(lo, hi, 0, 1, 2, 3, 4, 5, 6, 7,
                                 8, 9, 10, 11, 12, 13, 14, 15);
}

// B (32x16 f16) from LDS tile stored TRANSPOSED as [n][k]:
// lane L holds column N=L%16, K = kb..kb+7 and kb+16..kb+23 (kb=0 lanes<16,
// kb=8 otherwise). Two contiguous 8-half runs => 2x ds_load_b128.
__device__ __forceinline__ v16h frag_bt(const _Float16* sBt) {
  int lane = threadIdx.x & 31;
  const _Float16* row = sBt + (lane & 15) * LDB + ((lane >> 4) << 3);
  v8h lo = *(const v8h*)(row);
  v8h hi = *(const v8h*)(row + 16);
  return __builtin_shufflevector(lo, hi, 0, 1, 2, 3, 4, 5, 6, 7,
                                 8, 9, 10, 11, 12, 13, 14, 15);
}

__device__ __forceinline__ v8f wmma16(v16h a, v16h b, v8f c) {
  return __builtin_amdgcn_wmma_f32_16x16x32_f16(false, a, false, b, (short)0, c,
                                                false, false);
}

__device__ __forceinline__ float wave_sum32(float v) {
#pragma unroll
  for (int m = 16; m > 0; m >>= 1) v += __shfl_xor(v, m, 32);
  return v;
}

// ---------------- weight prep -------------------------------------------------
// smooth_w [3][3][384][256] f32 -> wt16 [tapch(54)][n(256)][k(64)] f16,
// tapch = tap*6 + ch, source channel = ch*64 + kk.
__global__ void prep_smooth_k(const float* __restrict__ src,
                              _Float16* __restrict__ wt16) {
  for (int i = blockIdx.x * blockDim.x + threadIdx.x; i < 884736;
       i += gridDim.x * blockDim.x) {
    int kk = i & 63;
    int n = (i >> 6) & 255;
    int tapch = i >> 14;                 // 0..53
    int tap = tapch / 6, ch = tapch % 6;
    wt16[i] = (_Float16)src[((size_t)(tap * CATCH + ch * 64 + kk)) * CCH + n];
  }
}

// dwn_w [16][256][8] f32 -> wdt16 [n(128)][k(256)] f16 (n = br*8+f); bd[128]
__global__ void prep_dwn_k(const float* __restrict__ dwn_w,
                           const float* __restrict__ dwn_b,
                           _Float16* __restrict__ wdt16, float* __restrict__ bd) {
  int i = blockIdx.x * 256 + threadIdx.x;
  if (i < 32768) {
    int col = i >> 8, c = i & 255;
    int br = col >> 3, f = col & 7;
    wdt16[i] = (_Float16)dwn_w[(br * 256 + c) * 8 + f];
  }
  if (i < 128) bd[i] = dwn_b[i];
}

// ---------------- input LayerNorm (wave per pixel) ----------------------------
__global__ void ln_in_k(const float* __restrict__ x, const float* __restrict__ g,
                        const float* __restrict__ b, _Float16* __restrict__ cat) {
  int wave = threadIdx.x >> 5, lane = threadIdx.x & 31;
  int pix = blockIdx.x * 8 + wave;
  const float4* p4 = (const float4*)(x + (size_t)pix * CCH + lane * 8);
  float4 a = p4[0], c4 = p4[1];
  float v[8] = {a.x, a.y, a.z, a.w, c4.x, c4.y, c4.z, c4.w};
  float s = 0.f, sq = 0.f;
#pragma unroll
  for (int i = 0; i < 8; ++i) { s += v[i]; sq += v[i] * v[i]; }
  s = wave_sum32(s);
  sq = wave_sum32(sq);
  float mean = s * (1.0f / 256.0f);
  float var = sq * (1.0f / 256.0f) - mean * mean;
  float inv = rsqrtf(var + EPSLN);
  _Float16 h[8];
#pragma unroll
  for (int i = 0; i < 8; ++i) {
    int ch = lane * 8 + i;
    h[i] = (_Float16)((v[i] - mean) * inv * g[ch] + b[ch]);
  }
  *(float4*)(cat + (size_t)pix * CATCH + lane * 8) = *(float4*)h;
}

// ---------------- shared 32K-slice MMA step (32x32 wave tile) -----------------
#define MMA_SLICE(ks)                                                          \
  {                                                                            \
    v16h a0 = frag_a(sA + (size_t)wm * LDA + (ks));                            \
    v16h a1 = frag_a(sA + (size_t)(wm + 16) * LDA + (ks));                     \
    v16h b0 = frag_bt(sBt + (size_t)wn * LDB + (ks));                          \
    v16h b1 = frag_bt(sBt + (size_t)(wn + 16) * LDB + (ks));                   \
    acc00 = wmma16(a0, b0, acc00);                                             \
    acc01 = wmma16(a0, b1, acc01);                                             \
    acc10 = wmma16(a1, b0, acc10);                                             \
    acc11 = wmma16(a1, b1, acc11);                                             \
  }

// ---------------- fused 1x1 down conv: GEMM M=65536 K=256 N=128 ---------------
// block: 64 pixels x 128 cols, 8 waves (2x4), wave tile 32x32, K chunks of 64.
__global__ void gemm_dwn_k(const _Float16* __restrict__ cat,
                           const _Float16* __restrict__ wdt16,
                           const float* __restrict__ bd, float* __restrict__ dx) {
  __shared__ _Float16 sA[64 * LDA];
  __shared__ _Float16 sBt[128 * LDB];
  int t = threadIdx.x;
  int mbase = blockIdx.x * 64;
  int wv = t >> 5, lane = t & 31;
  int wm = (wv >> 2) * 32, wn = (wv & 3) * 32;
  v8f acc00 = {}, acc01 = {}, acc10 = {}, acc11 = {};
  for (int cc = 0; cc < 256; cc += 64) {
#pragma unroll
    for (int r = 0; r < 2; ++r) {            // A: 64 rows x 64 halves
      int idx = t + r * 256;
      int p = idx >> 3, q = idx & 7;
      *(float4*)(sA + p * LDA + q * 8) =
          *(const float4*)(cat + (size_t)(mbase + p) * CATCH + cc + q * 8);
    }
#pragma unroll
    for (int r = 0; r < 4; ++r) {            // Bt: 128 n-rows x 64 halves
      int idx = t + r * 256;
      int n = idx >> 3, q = idx & 7;
      *(float4*)(sBt + n * LDB + q * 8) =
          *(const float4*)(wdt16 + (size_t)n * 256 + cc + q * 8);
    }
    __syncthreads();
    MMA_SLICE(0)
    MMA_SLICE(32)
    __syncthreads();
  }
  int lane15 = lane & 15, lhi = (lane >> 4) << 3;
#pragma unroll
  for (int r = 0; r < 8; ++r) {
    int m0 = mbase + wm + r + lhi, m1 = m0 + 16;
    int c0 = wn + lane15, c1 = c0 + 16;
    float u;
    u = acc00[r] + bd[c0]; dx[(size_t)m0 * 128 + c0] = u > 0.f ? u : 0.f;
    u = acc01[r] + bd[c1]; dx[(size_t)m0 * 128 + c1] = u > 0.f ? u : 0.f;
    u = acc10[r] + bd[c0]; dx[(size_t)m1 * 128 + c0] = u > 0.f ? u : 0.f;
    u = acc11[r] + bd[c1]; dx[(size_t)m1 * 128 + c1] = u > 0.f ? u : 0.f;
  }
}

// ---------------- per-branch k/q/v dilated 3x3 conv (8->8) --------------------
__global__ void kqv_k(const float* __restrict__ dx, const float* __restrict__ kw,
                      const float* __restrict__ qw, const float* __restrict__ vw,
                      const float* __restrict__ kb, const float* __restrict__ qb,
                      const float* __restrict__ vb, int br, int d,
                      float* __restrict__ kq, float* __restrict__ vout) {
  __shared__ float w[3 * 576];
  for (int i = threadIdx.x; i < 576; i += 256) {
    w[i] = kw[i];
    w[576 + i] = qw[i];
    w[1152 + i] = vw[i];
  }
  __syncthreads();
  int pix = blockIdx.x * 256 + threadIdx.x;
  int img = pix >> 14, sp = pix & 16383;
  int y = sp >> 7, x = sp & 127;
  float ak[8], aq[8], av[8];
#pragma unroll
  for (int co = 0; co < 8; ++co) { ak[co] = kb[co]; aq[co] = qb[co]; av[co] = vb[co]; }
  for (int ky = 0; ky < 3; ++ky) {
    int yy = y + (ky - 1) * d;
    if (yy < 0 || yy >= HH) continue;
    for (int kx = 0; kx < 3; ++kx) {
      int xx = x + (kx - 1) * d;
      if (xx < 0 || xx >= WW) continue;
      const float4* d4 =
          (const float4*)(dx + ((size_t)((img << 14) + (yy << 7) + xx)) * 128 + br * 8);
      float4 lo = d4[0], hi = d4[1];
      float di[8] = {lo.x, lo.y, lo.z, lo.w, hi.x, hi.y, hi.z, hi.w};
      int wb = (ky * 3 + kx) * 64;
#pragma unroll
      for (int ci = 0; ci < 8; ++ci)
#pragma unroll
        for (int co = 0; co < 8; ++co) {
          ak[co] += di[ci] * w[wb + ci * 8 + co];
          aq[co] += di[ci] * w[576 + wb + ci * 8 + co];
          av[co] += di[ci] * w[1152 + wb + ci * 8 + co];
        }
    }
  }
#pragma unroll
  for (int co = 0; co < 8; ++co) {
    float k = ak[co] > 0.f ? ak[co] : 0.f;
    float q = aq[co] > 0.f ? aq[co] : 0.f;
    float v = av[co] > 0.f ? av[co] : 0.f;
    kq[(size_t)pix * 8 + co] = k * q;
    vout[(size_t)pix * 8 + co] = v;
  }
}

// ---------------- spatial softmax stats: max & sum(exp) per (img, c) ----------
__global__ void softmax_stats_k(const float* __restrict__ kq,
                                float* __restrict__ stats) {
  __shared__ float red[256];
  int img = blockIdx.x >> 3, c = blockIdx.x & 7;
  const float* base = kq + ((size_t)img << 14) * 8 + c;
  float m = -1e30f;
  for (int sp = threadIdx.x; sp < HW; sp += 256)
    m = fmaxf(m, base[(size_t)sp * 8]);
  red[threadIdx.x] = m;
  __syncthreads();
  for (int s = 128; s > 0; s >>= 1) {
    if ((int)threadIdx.x < s)
      red[threadIdx.x] = fmaxf(red[threadIdx.x], red[threadIdx.x + s]);
    __syncthreads();
  }
  float gm = red[0];
  __syncthreads();
  float sum = 0.f;
  for (int sp = threadIdx.x; sp < HW; sp += 256)
    sum += __expf(base[(size_t)sp * 8] - gm);
  red[threadIdx.x] = sum;
  __syncthreads();
  for (int s = 128; s > 0; s >>= 1) {
    if ((int)threadIdx.x < s) red[threadIdx.x] += red[threadIdx.x + s];
    __syncthreads();
  }
  if (threadIdx.x == 0) {
    stats[blockIdx.x * 2] = gm;
    stats[blockIdx.x * 2 + 1] = red[0];
  }
}

// ---------------- attn*v + LN(8) -> cat channels 256+8*br ---------------------
__global__ void attn_fin_k(const float* __restrict__ kq, const float* __restrict__ vv,
                           const float* __restrict__ stats,
                           const float* __restrict__ g, const float* __restrict__ b,
                           int br, _Float16* __restrict__ cat) {
  int pix = blockIdx.x * 256 + threadIdx.x;
  int img = pix >> 14;
  float y[8], mean = 0.f;
#pragma unroll
  for (int c = 0; c < 8; ++c) {
    float mx = stats[(img * 8 + c) * 2];
    float sm = stats[(img * 8 + c) * 2 + 1];
    float a = __expf(kq[(size_t)pix * 8 + c] - mx) / sm;
    y[c] = a * vv[(size_t)pix * 8 + c];
    mean += y[c];
  }
  mean *= 0.125f;
  float var = 0.f;
#pragma unroll
  for (int c = 0; c < 8; ++c) { float d = y[c] - mean; var += d * d; }
  var *= 0.125f;
  float inv = rsqrtf(var + EPSLN);
  _Float16 h[8];
#pragma unroll
  for (int c = 0; c < 8; ++c) h[c] = (_Float16)((y[c] - mean) * inv * g[c] + b[c]);
  *(float4*)(cat + (size_t)pix * CATCH + 256 + br * 8) = *(float4*)h;
}

// ---------------- smooth conv: implicit GEMM M=65536 K=3456 N=256 -------------
// block: 64-pixel row strip x 128 out-channels; 54 staging rounds of K=64.
__global__ void gemm_smooth_k(const _Float16* __restrict__ cat,
                              const _Float16* __restrict__ wt16,
                              const float* __restrict__ sb, float* __restrict__ out) {
  __shared__ _Float16 sA[64 * LDA];
  __shared__ _Float16 sBt[128 * LDB];
  int t = threadIdx.x;
  int xs = (blockIdx.x & 1) * 64;
  int y = (blockIdx.x >> 1) & 127;
  int img = blockIdx.x >> 8;
  int nbase = blockIdx.y * 128;
  int wv = t >> 5, lane = t & 31;
  int wm = (wv >> 2) * 32, wn = (wv & 3) * 32;
  v8f acc00 = {}, acc01 = {}, acc10 = {}, acc11 = {};
  for (int tap = 0; tap < 9; ++tap) {
    int ky = tap / 3, kx = tap % 3;
    int yy = y + ky - 1;
    bool yok = (yy >= 0 && yy < HH);
    const _Float16* arow = cat + ((size_t)((img * HH + yy) * WW)) * CATCH;
    for (int ch = 0; ch < 6; ++ch) {
      int cc = ch * 64;
#pragma unroll
      for (int r = 0; r < 2; ++r) {          // A: 64 pixels x 64 halves
        int idx = t + r * 256;
        int p = idx >> 3, q = idx & 7;
        int xx = xs + p + kx - 1;
        float4 val = {0.f, 0.f, 0.f, 0.f};
        if (yok && xx >= 0 && xx < WW) {
          const _Float16* src = arow + (size_t)xx * CATCH + cc + q * 8;
          val = *(const float4*)src;
          __builtin_prefetch(src + 64, 0, 1);  // next K chunk
        }
        *(float4*)(sA + p * LDA + q * 8) = val;
      }
      const _Float16* wsrc =
          wt16 + ((size_t)((tap * 6 + ch) * 256 + nbase)) * 64;
#pragma unroll
      for (int r = 0; r < 4; ++r) {          // Bt: 128 n-rows x 64 halves
        int idx = t + r * 256;
        int n = idx >> 3, q = idx & 7;
        *(float4*)(sBt + n * LDB + q * 8) =
            *(const float4*)(wsrc + (size_t)n * 64 + q * 8);
      }
      __syncthreads();
      MMA_SLICE(0)
      MMA_SLICE(32)
      __syncthreads();
    }
  }
  int lane15 = lane & 15, lhi = (lane >> 4) << 3;
  size_t rowbase = ((size_t)((img * HH + y) * WW)) * CCH;
#pragma unroll
  for (int r = 0; r < 8; ++r) {
    int x0 = xs + wm + r + lhi, x1 = x0 + 16;
    int c0 = nbase + wn + lane15, c1 = c0 + 16;
    float u;
    u = acc00[r] + sb[c0]; out[rowbase + (size_t)x0 * CCH + c0] = u > 0.f ? u : 0.f;
    u = acc01[r] + sb[c1]; out[rowbase + (size_t)x0 * CCH + c1] = u > 0.f ? u : 0.f;
    u = acc10[r] + sb[c0]; out[rowbase + (size_t)x1 * CCH + c0] = u > 0.f ? u : 0.f;
    u = acc11[r] + sb[c1]; out[rowbase + (size_t)x1 * CCH + c1] = u > 0.f ? u : 0.f;
  }
}

// ---------------- output LayerNorm (in-place on d_out) ------------------------
__global__ void ln_out_k(float* __restrict__ out, const float* __restrict__ g,
                         const float* __restrict__ b) {
  int wave = threadIdx.x >> 5, lane = threadIdx.x & 31;
  int pix = blockIdx.x * 8 + wave;
  float* px = out + (size_t)pix * CCH + lane * 8;
  float4 a = ((float4*)px)[0], c4 = ((float4*)px)[1];
  float v[8] = {a.x, a.y, a.z, a.w, c4.x, c4.y, c4.z, c4.w};
  float s = 0.f, sq = 0.f;
#pragma unroll
  for (int i = 0; i < 8; ++i) { s += v[i]; sq += v[i] * v[i]; }
  s = wave_sum32(s);
  sq = wave_sum32(sq);
  float mean = s * (1.0f / 256.0f);
  float var = sq * (1.0f / 256.0f) - mean * mean;
  float inv = rsqrtf(var + EPSLN);
  float r[8];
#pragma unroll
  for (int i = 0; i < 8; ++i) {
    int ch = lane * 8 + i;
    r[i] = (v[i] - mean) * inv * g[ch] + b[ch];
  }
  ((float4*)px)[0] = make_float4(r[0], r[1], r[2], r[3]);
  ((float4*)px)[1] = make_float4(r[4], r[5], r[6], r[7]);
}

// ---------------- host launcher ------------------------------------------------
extern "C" void kernel_launch(void* const* d_in, const int* in_sizes, int n_in,
                              void* d_out, int out_size, void* d_ws, size_t ws_size,
                              hipStream_t stream) {
  (void)in_sizes; (void)n_in; (void)out_size;
  const float* x        = (const float*)d_in[0];
  const float* dwn_w    = (const float*)d_in[1];
  const float* dwn_b    = (const float*)d_in[2];
  const float* k_w      = (const float*)d_in[3];
  const float* k_b      = (const float*)d_in[4];
  const float* q_w      = (const float*)d_in[5];
  const float* q_b      = (const float*)d_in[6];
  const float* v_w      = (const float*)d_in[7];
  const float* v_b      = (const float*)d_in[8];
  const float* smooth_w = (const float*)d_in[9];
  const float* smooth_b = (const float*)d_in[10];
  const float* ln_in_g  = (const float*)d_in[11];
  const float* ln_in_b  = (const float*)d_in[12];
  const float* ln_up_g  = (const float*)d_in[13];
  const float* ln_up_b  = (const float*)d_in[14];
  const float* ln_out_g = (const float*)d_in[15];
  const float* ln_out_b = (const float*)d_in[16];
  float* out = (float*)d_out;

  char* ws = (char*)d_ws;
  size_t off = 0;
  _Float16* cat  = (_Float16*)(ws + off); off += (size_t)NPIX * CATCH * 2;  // 48 MB
  float* dx      = (float*)(ws + off);    off += (size_t)NPIX * 128 * 4;    // 32 MB
  float* kq      = (float*)(ws + off);    off += (size_t)NPIX * 8 * 4;      // 2 MB
  float* vv      = (float*)(ws + off);    off += (size_t)NPIX * 8 * 4;      // 2 MB
  _Float16* wt16 = (_Float16*)(ws + off); off += (size_t)884736 * 2;
  _Float16* wdt16= (_Float16*)(ws + off); off += (size_t)32768 * 2;
  float* bd      = (float*)(ws + off);    off += 128 * 4;
  float* stats   = (float*)(ws + off);    off += 64 * 4;
  if (ws_size < off) return;

  prep_smooth_k<<<1024, 256, 0, stream>>>(smooth_w, wt16);
  prep_dwn_k<<<128, 256, 0, stream>>>(dwn_w, dwn_b, wdt16, bd);
  ln_in_k<<<NPIX / 8, 256, 0, stream>>>(x, ln_in_g, ln_in_b, cat);
  gemm_dwn_k<<<NPIX / 64, 256, 0, stream>>>(cat, wdt16, bd, dx);
  for (int br = 0; br < NDB; ++br) {
    kqv_k<<<NPIX / 256, 256, 0, stream>>>(dx, k_w + br * 576, q_w + br * 576,
                                          v_w + br * 576, k_b + br * 8,
                                          q_b + br * 8, v_b + br * 8, br, br + 1,
                                          kq, vv);
    softmax_stats_k<<<32, 256, 0, stream>>>(kq, stats);
    attn_fin_k<<<NPIX / 256, 256, 0, stream>>>(kq, vv, stats, ln_up_g, ln_up_b,
                                               br, cat);
  }
  gemm_smooth_k<<<dim3(1024, 2), 256, 0, stream>>>(cat, wt16, smooth_b, out);
  ln_out_k<<<NPIX / 8, 256, 0, stream>>>(out, ln_out_g, ln_out_b);
}
